// SampledSoftmax_18811956756725
// MI455X (gfx1250) — matrix operand order repaired and checked by reading, not compiled, not run on hardware
//
#include <hip/hip_runtime.h>

// ---------------- problem constants ----------------
#define NB      8        // batch
#define NT      2048     // timesteps
#define ND      1024     // d_model (GEMM K)
#define NS      1000     // sampled
#define NSPAD   1024     // padded sampled columns
#define NCLS    50000

typedef __attribute__((ext_vector_type(16))) __bf16 v16bf;
typedef __attribute__((ext_vector_type(8)))  __bf16 v8bf;
typedef __attribute__((ext_vector_type(4)))  __bf16 v4bf;
typedef __attribute__((ext_vector_type(8)))  float  v8f;

union V16U { v16bf v; v8bf h[2]; };

// TF log-uniform expected-count correction, log(-expm1(K*log1p(-p)))
__device__ __forceinline__ float log_expected(int id) {
    float c = (float)id;
    float p = log1pf(1.0f / (c + 1.0f)) * (1.0f / logf((float)(NCLS + 1)));
    return logf(-expm1f((float)NS * log1pf(-p)));
}

// ---------------------------------------------------------------------------
// Kernel 1: gather W[sampled_ids] per batch, split fp32 -> bf16 hi + bf16 lo,
// pad columns [1000,1024) with zeros / bias -1e9. One block per (b, kcol).
// ---------------------------------------------------------------------------
__global__ __launch_bounds__(128)
void k_gather_split(const int* __restrict__ sid, const float* __restrict__ W,
                    const float* __restrict__ bias,
                    __bf16* __restrict__ whi, __bf16* __restrict__ wlo,
                    float* __restrict__ sbw) {
    int blk = blockIdx.x;                 // b*NSPAD + k
    int b = blk >> 10, k = blk & (NSPAD - 1);
    size_t rowOut = ((size_t)blk) << 10;  // *ND
    int tid = threadIdx.x;
    if (k < NS) {
        int id = sid[b * NS + k];
        const float* wr = W + ((size_t)id << 10);
        for (int d = tid * 4; d < ND; d += 512) {
            float4 x = *(const float4*)(wr + d);
            __bf16 h0 = (__bf16)x.x; __bf16 l0 = (__bf16)(x.x - (float)h0);
            __bf16 h1 = (__bf16)x.y; __bf16 l1 = (__bf16)(x.y - (float)h1);
            __bf16 h2 = (__bf16)x.z; __bf16 l2 = (__bf16)(x.z - (float)h2);
            __bf16 h3 = (__bf16)x.w; __bf16 l3 = (__bf16)(x.w - (float)h3);
            v4bf hv = {h0, h1, h2, h3};
            v4bf lv = {l0, l1, l2, l3};
            *(v4bf*)(whi + rowOut + d) = hv;
            *(v4bf*)(wlo + rowOut + d) = lv;
        }
        if (tid == 0) sbw[blk] = bias[id] - log_expected(id);
    } else {
        v4bf z = {(__bf16)0.f, (__bf16)0.f, (__bf16)0.f, (__bf16)0.f};
        for (int d = tid * 4; d < ND; d += 512) {
            *(v4bf*)(whi + rowOut + d) = z;
            *(v4bf*)(wlo + rowOut + d) = z;
        }
        if (tid == 0) sbw[blk] = -1.0e9f;
    }
}

// ---------------------------------------------------------------------------
// Kernel 2: fused GEMM (bf16x3 WMMA) + streaming softmax per 32-row t-tile.
// Grid: (NT/32, NB), block: 256 threads = 8 waves (2 M-waves x 4 N-waves).
// ---------------------------------------------------------------------------
#define ASTR 1032   // padded LDS row stride in bf16 elements (conflict-free)

__global__ __launch_bounds__(256)
void k_sampled_softmax(const float* __restrict__ h, const int* __restrict__ nid,
                       const int* __restrict__ sid, const float* __restrict__ W,
                       const float* __restrict__ bias,
                       const __bf16* __restrict__ whi, const __bf16* __restrict__ wlo,
                       const float* __restrict__ sbw, float* __restrict__ lossRow) {
    __shared__ __bf16 sAhi[32 * ASTR];
    __shared__ __bf16 sAlo[32 * ASTR];
    __shared__ float  sSb[NSPAD];
    __shared__ int    sSid[NSPAD];
    __shared__ int    sNid[32];
    __shared__ float  sTl[32];
    __shared__ float  sChunk[32 * 65];

    const int b   = blockIdx.y;
    const int t0  = blockIdx.x * 32;
    const int tid = threadIdx.x;
    const int lane = tid & 31;
    const int wave = tid >> 5;
    const int mw = wave & 1;      // 0..1 : M half
    const int nw = wave >> 1;     // 0..3 : N quarter

    // ---- stage h tile into LDS as bf16 hi/lo (32 rows x 1024) ----
    for (int i = tid; i < 8192; i += 256) {
        int r = i >> 8;
        int d = (i & 255) * 4;
        float4 x = *(const float4*)(h + (((size_t)b * NT + t0 + r) << 10) + d);
        __bf16 h0 = (__bf16)x.x; __bf16 l0 = (__bf16)(x.x - (float)h0);
        __bf16 h1 = (__bf16)x.y; __bf16 l1 = (__bf16)(x.y - (float)h1);
        __bf16 h2 = (__bf16)x.z; __bf16 l2 = (__bf16)(x.z - (float)h2);
        __bf16 h3 = (__bf16)x.w; __bf16 l3 = (__bf16)(x.w - (float)h3);
        v4bf hv = {h0, h1, h2, h3};
        v4bf lv = {l0, l1, l2, l3};
        *(v4bf*)(&sAhi[r * ASTR + d]) = hv;
        *(v4bf*)(&sAlo[r * ASTR + d]) = lv;
    }
    for (int i = tid; i < NSPAD; i += 256) {
        sSb[i]  = sbw[b * NSPAD + i];
        sSid[i] = (i < NS) ? sid[b * NS + i] : -1;
    }
    if (tid < 32) sNid[tid] = nid[b * NT + t0 + tid];

    // ---- true logit: 8 lanes per row, fp32 ----
    {
        int t = tid >> 3, j = tid & 7;
        int id = nid[b * NT + t0 + t];
        const float* hr = h + (((size_t)b * NT + t0 + t) << 10) + j * 128;
        const float* wr = W + ((size_t)id << 10) + j * 128;
        float s = 0.f;
        for (int d = 0; d < 128; d += 4) {
            float4 a = *(const float4*)(hr + d);
            float4 w = *(const float4*)(wr + d);
            s += a.x * w.x + a.y * w.y + a.z * w.z + a.w * w.w;
        }
        s += __shfl_xor(s, 1, 32);
        s += __shfl_xor(s, 2, 32);
        s += __shfl_xor(s, 4, 32);
        if (j == 0) sTl[t] = s + bias[id] - log_expected(id);
    }
    __syncthreads();

    // ---- WMMA fragment addressing ----
    const int rowA = mw * 16 + (lane & 15);
    const int sel  = lane >> 4;
    const int colL = nw * 16 + (lane & 15);      // column within 64-wide chunk
    const __bf16* ahiP = &sAhi[rowA * ASTR + sel * 16];
    const __bf16* aloP = &sAlo[rowA * ASTR + sel * 16];

    float m_run = -3.0e38f, s_run = 0.f;

    for (int nc = 0; nc < NSPAD / 64; ++nc) {
        const int nbase = nc * 64;
        const size_t brow = ((size_t)(b * NSPAD + nbase + colL)) << 10;
        const __bf16* bhiP = whi + brow + sel * 16;
        const __bf16* bloP = wlo + brow + sel * 16;

        v8f acc = {0.f, 0.f, 0.f, 0.f, 0.f, 0.f, 0.f, 0.f};
        #pragma unroll 4
        for (int kk = 0; kk < ND; kk += 32) {
            // Unconditional speculative prefetch: can run past the end of the 16 MiB
            // panel; invalid/duplicate lines are dropped by HW. No branch.
            __builtin_prefetch(bhiP + kk + 256, 0, 1);
            __builtin_prefetch(bloP + kk + 256, 0, 1);
            V16U ah, al;
            ah.h[0] = *(const v8bf*)(ahiP + kk);
            ah.h[1] = *(const v8bf*)(ahiP + kk + 8);
            al.h[0] = *(const v8bf*)(aloP + kk);
            al.h[1] = *(const v8bf*)(aloP + kk + 8);
            v16bf bh = *(const v16bf*)(bhiP + kk);
            v16bf bl = *(const v16bf*)(bloP + kk);
            acc = __builtin_amdgcn_wmma_f32_16x16x32_bf16(false, ah.v, false, bh,
                                                          (short)0, acc, false, false);
            acc = __builtin_amdgcn_wmma_f32_16x16x32_bf16(false, ah.v, false, bl,
                                                          (short)0, acc, false, false);
            acc = __builtin_amdgcn_wmma_f32_16x16x32_bf16(false, al.v, false, bh,
                                                          (short)0, acc, false, false);
        }

        // bias + accidental-hit mask, spill 32x64 chunk to LDS
        const float sbv = sSb[nbase + colL];
        const int   sv  = sSid[nbase + colL];
        const int rbase = mw * 16 + sel * 8;
        #pragma unroll
        for (int v = 0; v < 8; ++v) {
            int rr = rbase + v;
            float val = acc[v] + sbv;
            if (sv == sNid[rr]) val = -1.0e9f;
            sChunk[rr * 65 + colL] = val;
        }
        __syncthreads();

        // wave-0 online softmax update: one thread per row
        if (tid < 32) {
            const float* rowp = &sChunk[tid * 65];
            float mc = -3.0e38f;
            #pragma unroll 8
            for (int j = 0; j < 64; ++j) mc = fmaxf(mc, rowp[j]);
            float nm = fmaxf(m_run, mc);
            float s = s_run * __expf(m_run - nm);
            #pragma unroll 8
            for (int j = 0; j < 64; ++j) s += __expf(rowp[j] - nm);
            s_run = s;
            m_run = nm;
        }
        __syncthreads();
    }

    if (tid < 32) {
        float tl = sTl[tid];
        float nm  = fmaxf(m_run, tl);
        float tot = s_run * __expf(m_run - nm) + __expf(tl - nm);
        lossRow[b * NT + t0 + tid] = nm + logf(tot) - tl;   // -log_softmax[0]
    }
}

// ---------------------------------------------------------------------------
// Kernel 3: mean over T per batch
// ---------------------------------------------------------------------------
__global__ __launch_bounds__(256)
void k_reduce_mean(const float* __restrict__ lossRow, float* __restrict__ out) {
    __shared__ float red[256];
    int b = blockIdx.x, tid = threadIdx.x;
    float s = 0.f;
    for (int i = tid; i < NT; i += 256) s += lossRow[b * NT + i];
    red[tid] = s;
    __syncthreads();
    for (int off = 128; off > 0; off >>= 1) {
        if (tid < off) red[tid] += red[tid + off];
        __syncthreads();
    }
    if (tid == 0) out[b] = red[0] * (1.0f / (float)NT);
}

// ---------------------------------------------------------------------------
extern "C" void kernel_launch(void* const* d_in, const int* in_sizes, int n_in,
                              void* d_out, int out_size, void* d_ws, size_t ws_size,
                              hipStream_t stream) {
    const float* h    = (const float*)d_in[0];   // [B,T,D]
    const int*   nid  = (const int*)  d_in[1];   // [B,T]
    const int*   sidp = (const int*)  d_in[2];   // [B,NS]
    const float* W    = (const float*)d_in[3];   // [NCLS,D]
    const float* bias = (const float*)d_in[4];   // [NCLS]

    char* ws = (char*)d_ws;
    const size_t WPAN = (size_t)NB * NSPAD * ND * sizeof(__bf16);   // 16 MiB
    __bf16* whi     = (__bf16*)(ws);
    __bf16* wlo     = (__bf16*)(ws + WPAN);
    float*  sbw     = (float*)(ws + 2 * WPAN);                       // [B,NSPAD]
    float*  lossRow = (float*)(ws + 2 * WPAN + (size_t)NB * NSPAD * 4);

    k_gather_split<<<NB * NSPAD, 128, 0, stream>>>(sidp, W, bias, whi, wlo, sbw);

    dim3 g2(NT / 32, NB);
    k_sampled_softmax<<<g2, 256, 0, stream>>>(h, nid, sidp, W, bias,
                                              whi, wlo, sbw, lossRow);

    k_reduce_mean<<<NB, 256, 0, stream>>>(lossRow, (float*)d_out);
}
